// CTC_30906584662757
// MI455X (gfx1250) — matrix operand order repaired and checked by reading, not compiled, not run on hardware
//
#include <hip/hip_runtime.h>
#include <hip/hip_bf16.h>
#include <stddef.h>

// Problem constants (from reference setup_inputs)
#define BB   16
#define TT   800
#define DD   512
#define VV   5000
#define LL   100
#define SS   201          // 2L+1
#define VP   5120         // V padded to 320 tiles (8 waves x 2 tiles x 20 iters)
#define NTIL 320          // VP/16
#define NJ   101          // blank + L labels
#define JP   112          // NJ padded to 16 (7 tiles)
#define NEGF (-1e30f)

typedef __attribute__((ext_vector_type(16))) __bf16 v16bf;
typedef __attribute__((ext_vector_type(8)))  __bf16 bf16x8;
typedef __attribute__((ext_vector_type(8)))  float  v8f;

union Frag { v16bf v; bf16x8 p[2]; };

static __device__ __forceinline__ v8f wmma_bf16(v16bf a, v16bf b, v8f c) {
    return __builtin_amdgcn_wmma_f32_16x16x32_bf16(
        /*neg_a=*/false, a, /*neg_b=*/false, b,
        /*c_mod=*/(short)0, c, /*reuse_a=*/false, /*reuse_b=*/false);
}

// ---------------- split-bf16 conversion kernels ----------------

__global__ void cvt_split(const float* __restrict__ x,
                          __bf16* __restrict__ hi, __bf16* __restrict__ lo, int n) {
    for (int i = blockIdx.x * blockDim.x + threadIdx.x; i < n;
         i += gridDim.x * blockDim.x) {
        float v = x[i];
        __bf16 h = (__bf16)v;
        hi[i] = h;
        lo[i] = (__bf16)(v - (float)h);
    }
}

__global__ void cvt_w_pad(const float* __restrict__ w,
                          __bf16* __restrict__ hi, __bf16* __restrict__ lo) {
    const int nValid = VV * DD, nTot = VP * DD;
    for (int i = blockIdx.x * blockDim.x + threadIdx.x; i < nTot;
         i += gridDim.x * blockDim.x) {
        if (i < nValid) {
            float v = w[i];
            __bf16 h = (__bf16)v;
            hi[i] = h;
            lo[i] = (__bf16)(v - (float)h);
        } else {
            hi[i] = (__bf16)0.0f;
            lo[i] = (__bf16)0.0f;
        }
    }
}

// Build gathered W rows per sample: j=0 -> blank(0), j=1..100 -> labels, j>=101 -> zero
__global__ void gather_w(const __bf16* __restrict__ wHi, const __bf16* __restrict__ wLo,
                         const float* __restrict__ bias, const int* __restrict__ ys,
                         __bf16* __restrict__ gHi, __bf16* __restrict__ gLo,
                         float* __restrict__ bg) {
    int blk = blockIdx.x;            // 0 .. B*JP-1
    int b = blk / JP, j = blk % JP;
    int id = (j == 0) ? 0 : ((j < NJ) ? ys[b * LL + (j - 1)] : -1);
    __bf16* dH = gHi + ((size_t)(b * JP + j)) * DD;
    __bf16* dL = gLo + ((size_t)(b * JP + j)) * DD;
    if (id >= 0) {
        const __bf16* sH = wHi + (size_t)id * DD;
        const __bf16* sL = wLo + (size_t)id * DD;
        for (int k = threadIdx.x; k < DD; k += blockDim.x) { dH[k] = sH[k]; dL[k] = sL[k]; }
        if (threadIdx.x == 0) bg[b * JP + j] = bias[id];
    } else {
        for (int k = threadIdx.x; k < DD; k += blockDim.x) {
            dH[k] = (__bf16)0.0f; dL[k] = (__bf16)0.0f;
        }
        if (threadIdx.x == 0) bg[b * JP + j] = 0.0f;
    }
}

// ---------------- main GEMM + fused row-wise logsumexp ----------------
// One block per 16 rows of the flat (B*T, D) activation matrix; 8 waves
// stripe the 320 N-tiles in pairs. Split-bf16: C = aH*bH + aH*bL + aL*bH,
// kept as two independent accumulator chains per tile so consecutive WMMAs
// never RAW-depend on each other (4 chains in flight per wave).
__global__ __launch_bounds__(256) void gemm_lse(
    const __bf16* __restrict__ hsHi, const __bf16* __restrict__ hsLo,
    const __bf16* __restrict__ wHi,  const __bf16* __restrict__ wLo,
    const float* __restrict__ bias,  float* __restrict__ lse) {
    int mt = blockIdx.x;             // 0..799
    int mbase = mt * 16;
    int tid = threadIdx.x;
    int wave = tid >> 5;
    int lane = tid & 31;
    int l = lane & 15, hb = lane >> 4;

    const __bf16* aHiRow = hsHi + (size_t)(mbase + l) * DD;
    const __bf16* aLoRow = hsLo + (size_t)(mbase + l) * DD;

    float mrun[8], srun[8];
#pragma unroll
    for (int r = 0; r < 8; ++r) { mrun[r] = NEGF; srun[r] = 0.0f; }

    for (int nt0 = wave * 2; nt0 < NTIL; nt0 += 16) {
        int vcol0 = nt0 * 16 + l;
        int vcol1 = vcol0 + 16;      // tile nt0+1, always valid memory (VP padded)
        const __bf16* b0HiRow = wHi + (size_t)vcol0 * DD;
        const __bf16* b0LoRow = wLo + (size_t)vcol0 * DD;
        const __bf16* b1HiRow = wHi + (size_t)vcol1 * DD;
        const __bf16* b1LoRow = wLo + (size_t)vcol1 * DD;
        if (nt0 + 16 < NTIL)  // warm caches for the next pair this wave owns
            __builtin_prefetch(wHi + (size_t)(vcol0 + 256) * DD, 0, 1);

        v8f c0a = {}, c0b = {}, c1a = {}, c1b = {};
        for (int k0 = 0; k0 < DD; k0 += 32) {
            Frag aH, aL, b0H, b0L, b1H, b1L;
            int ao = k0 + hb * 16;    // A: 16 contiguous K per lane-half
            int bo = k0 + hb * 16;    // B: 16 contiguous K per lane-half
            aH.p[0]  = *(const bf16x8*)(aHiRow + ao);
            aH.p[1]  = *(const bf16x8*)(aHiRow + ao + 8);
            aL.p[0]  = *(const bf16x8*)(aLoRow + ao);
            aL.p[1]  = *(const bf16x8*)(aLoRow + ao + 8);
            b0H.p[0] = *(const bf16x8*)(b0HiRow + bo);
            b0H.p[1] = *(const bf16x8*)(b0HiRow + bo + 8);
            b0L.p[0] = *(const bf16x8*)(b0LoRow + bo);
            b0L.p[1] = *(const bf16x8*)(b0LoRow + bo + 8);
            b1H.p[0] = *(const bf16x8*)(b1HiRow + bo);
            b1H.p[1] = *(const bf16x8*)(b1HiRow + bo + 8);
            b1L.p[0] = *(const bf16x8*)(b1LoRow + bo);
            b1L.p[1] = *(const bf16x8*)(b1LoRow + bo + 8);
            // 4 independent accumulation chains -> no WMMA->WMMA RAW stalls
            c0a = wmma_bf16(aH.v, b0H.v, c0a);
            c1a = wmma_bf16(aH.v, b1H.v, c1a);
            c0b = wmma_bf16(aH.v, b0L.v, c0b);
            c1b = wmma_bf16(aH.v, b1L.v, c1b);
            c0b = wmma_bf16(aL.v, b0H.v, c0b);
            c1b = wmma_bf16(aL.v, b1H.v, c1b);
        }
        bool v0 = vcol0 < VV;
        bool v1 = vcol1 < VV;
        float bv0 = v0 ? bias[vcol0] : 0.0f;
        float bv1 = v1 ? bias[vcol1] : 0.0f;
#pragma unroll
        for (int r = 0; r < 8; ++r) {
            float x0 = v0 ? (c0a[r] + c0b[r] + bv0) : NEGF;
            float x1 = v1 ? (c1a[r] + c1b[r] + bv1) : NEGF;
            if (x0 > mrun[r]) { srun[r] = srun[r] * __expf(mrun[r] - x0) + 1.0f; mrun[r] = x0; }
            else              { srun[r] += __expf(x0 - mrun[r]); }
            if (x1 > mrun[r]) { srun[r] = srun[r] * __expf(mrun[r] - x1) + 1.0f; mrun[r] = x1; }
            else              { srun[r] += __expf(x1 - mrun[r]); }
        }
    }

    __shared__ float pm[16][129];
    __shared__ float ps[16][129];
    int slot = wave * 16 + l;        // 0..127
#pragma unroll
    for (int r = 0; r < 8; ++r) {
        int row = r + hb * 8;
        pm[row][slot] = mrun[r];
        ps[row][slot] = srun[r];
    }
    __syncthreads();
    if (tid < 16) {
        float M = NEGF, S = 0.0f;
        for (int i = 0; i < 128; ++i) {
            float mi = pm[tid][i], si = ps[tid][i];
            if (mi > M) { S = S * __expf(M - mi) + si; M = mi; }
            else        { S += si * __expf(mi - M); }
        }
        lse[mbase + tid] = M + __logf(S);
    }
}

// ---------------- gathered-label GEMM: glog[b,t,j] = hs[b,t,:].Wg[b,j,:] + bg ----
__global__ __launch_bounds__(224) void gemm_gather(
    const __bf16* __restrict__ hsHi, const __bf16* __restrict__ hsLo,
    const __bf16* __restrict__ gHi,  const __bf16* __restrict__ gLo,
    const float* __restrict__ bg,    float* __restrict__ glog) {
    int mt = blockIdx.x;             // 0..49 (t tiles)
    int b  = blockIdx.y;
    int tid = threadIdx.x;
    int wave = tid >> 5;             // 0..6 -> one N-tile each
    int lane = tid & 31;
    int l = lane & 15, hb = lane >> 4;

    const __bf16* aHiRow = hsHi + (size_t)(b * TT + mt * 16 + l) * DD;
    const __bf16* aLoRow = hsLo + (size_t)(b * TT + mt * 16 + l) * DD;
    int jcol = wave * 16 + l;        // < 112 always
    const __bf16* bHiRow = gHi + (size_t)(b * JP + jcol) * DD;
    const __bf16* bLoRow = gLo + (size_t)(b * JP + jcol) * DD;

    v8f ca = {}, cb = {};            // two independent chains
    for (int k0 = 0; k0 < DD; k0 += 32) {
        Frag aH, aL, bH, bL;
        int ao = k0 + hb * 16;
        int bo = k0 + hb * 16;
        aH.p[0] = *(const bf16x8*)(aHiRow + ao);
        aH.p[1] = *(const bf16x8*)(aHiRow + ao + 8);
        aL.p[0] = *(const bf16x8*)(aLoRow + ao);
        aL.p[1] = *(const bf16x8*)(aLoRow + ao + 8);
        bH.p[0] = *(const bf16x8*)(bHiRow + bo);
        bH.p[1] = *(const bf16x8*)(bHiRow + bo + 8);
        bL.p[0] = *(const bf16x8*)(bLoRow + bo);
        bL.p[1] = *(const bf16x8*)(bLoRow + bo + 8);
        ca = wmma_bf16(aH.v, bH.v, ca);
        cb = wmma_bf16(aH.v, bL.v, cb);
        cb = wmma_bf16(aL.v, bH.v, cb);
    }
    float bv = bg[b * JP + jcol];
#pragma unroll
    for (int r = 0; r < 8; ++r) {
        int t = mt * 16 + r + hb * 8;
        glog[((size_t)(b * TT + t)) * JP + jcol] = ca[r] + cb[r] + bv;
    }
}

// ---------------- sequential CTC forward scan (one block per sample) ----------
__global__ __launch_bounds__(256) void ctc_scan(
    const float* __restrict__ glog, const float* __restrict__ lse,
    const int* __restrict__ ys, const int* __restrict__ hlens,
    const int* __restrict__ olens, float* __restrict__ losses) {
    int b = blockIdx.x;
    int s = threadIdx.x;             // 0..255, active for s < SS

    int ext = 0, ext2 = 0, j = 0;
    bool allow = false;
    if (s < SS) {
        if (s & 1) ext = ys[b * LL + (s >> 1)];
        if ((s >= 2) && (s & 1)) ext2 = ys[b * LL + (s >> 1) - 1];
        allow = (s >= 2) && (ext != 0) && (ext != ext2);
        j = (s & 1) ? ((s >> 1) + 1) : 0;
    }

    __shared__ float buf[2][SS + 3];
    if (s < SS) buf[0][s] = (s == 0) ? 0.0f : NEGF;
    __syncthreads();

    int hlen = hlens[b];
    int olen = olens[b];
    int cur = 0;
    for (int t = 0; t < hlen; ++t) {
        if (s < SS) {
            float lp = glog[((size_t)(b * TT + t)) * JP + j] - lse[b * TT + t];
            float a1 = buf[cur][s];
            float a2 = (s >= 1) ? buf[cur][s - 1] : NEGF;
            float a3 = allow ? buf[cur][s - 2] : NEGF;
            float m = fmaxf(a1, fmaxf(a2, a3));
            buf[cur ^ 1][s] =
                m + __logf(__expf(a1 - m) + __expf(a2 - m) + __expf(a3 - m)) + lp;
        }
        __syncthreads();
        cur ^= 1;
    }
    if (s == 0) {
        float v1 = buf[cur][2 * olen];
        float v2 = buf[cur][2 * olen - 1];
        float m = fmaxf(v1, v2);
        losses[b] = -(m + __logf(__expf(v1 - m) + __expf(v2 - m)));
    }
}

__global__ void final_reduce(const float* __restrict__ losses, float* __restrict__ out) {
    if (threadIdx.x == 0) {
        float s = 0.0f;
        for (int i = 0; i < BB; ++i) s += losses[i];
        out[0] = s / (float)BB;
    }
}

// ---------------- host launch ----------------

extern "C" void kernel_launch(void* const* d_in, const int* in_sizes, int n_in,
                              void* d_out, int out_size, void* d_ws, size_t ws_size,
                              hipStream_t stream) {
    const float* hs    = (const float*)d_in[0];   // [B,T,D]
    const int*   hlens = (const int*)d_in[1];     // [B]
    const int*   ys    = (const int*)d_in[2];     // [B,L]
    const int*   olens = (const int*)d_in[3];     // [B]
    const float* W     = (const float*)d_in[4];   // [V,D]
    const float* bias  = (const float*)d_in[5];   // [V]
    float* out = (float*)d_out;

    char* ws = (char*)d_ws;
    size_t off = 0;
    auto carve = [&](size_t bytes) -> char* {
        char* p = ws + off;
        off += (bytes + 255) & ~(size_t)255;
        return p;
    };
    const size_t nHS = (size_t)BB * TT * DD;
    __bf16* hsHi = (__bf16*)carve(nHS * 2);
    __bf16* hsLo = (__bf16*)carve(nHS * 2);
    __bf16* wHi  = (__bf16*)carve((size_t)VP * DD * 2);
    __bf16* wLo  = (__bf16*)carve((size_t)VP * DD * 2);
    __bf16* gHi  = (__bf16*)carve((size_t)BB * JP * DD * 2);
    __bf16* gLo  = (__bf16*)carve((size_t)BB * JP * DD * 2);
    float*  bg   = (float*)carve((size_t)BB * JP * 4);
    float*  lseb = (float*)carve((size_t)BB * TT * 4);
    float*  glog = (float*)carve((size_t)BB * TT * JP * 4);
    float*  loss = (float*)carve((size_t)BB * 4);

    cvt_split<<<4096, 256, 0, stream>>>(hs, hsHi, hsLo, (int)nHS);
    cvt_w_pad<<<4096, 256, 0, stream>>>(W, wHi, wLo);
    gather_w<<<BB * JP, 128, 0, stream>>>(wHi, wLo, bias, ys, gHi, gLo, bg);
    gemm_lse<<<(BB * TT) / 16, 256, 0, stream>>>(hsHi, hsLo, wHi, wLo, bias, lseb);
    gemm_gather<<<dim3(TT / 16, BB), 224, 0, stream>>>(hsHi, hsLo, gHi, gLo, bg, glog);
    ctc_scan<<<BB, 256, 0, stream>>>(glog, lseb, ys, hlens, olens, loss);
    final_reduce<<<1, 32, 0, stream>>>(loss, out);
}
